// HierarchicalStateSpaceModel_47717086659046
// MI455X (gfx1250) — compile-verified, matrix operand order checked by reading, not run on hardware
//
#include <hip/hip_runtime.h>
#include <hip/hip_bf16.h>

typedef _Float16 h16;
typedef __attribute__((ext_vector_type(16))) _Float16 v16h;
typedef __attribute__((ext_vector_type(8)))  _Float16 v8h;
typedef __attribute__((ext_vector_type(8)))  float    v8f;
typedef __attribute__((ext_vector_type(4)))  float    v4f;
typedef __attribute__((ext_vector_type(4)))  unsigned int u32x4;
typedef __attribute__((ext_vector_type(8)))  int      i32x8;
typedef __attribute__((ext_vector_type(4)))  int      i32x4;

#define DEVI __device__ __forceinline__

#define BATCH 64
#define TT    1000
#define TPN   998
#define HD    128
#define SD    32
#define KCB   10
#define BT    64000

// ---- workspace float offsets (time-major activation buffers [t][b][128]) ----
#define WS_ENCACT 0u
#define WS_ENCOBS 8192000u
#define WS_COMB   16384000u
#define WS_CBUFA  24576000u
#define WS_CBUFB  32768000u
#define WS_ABSFWD 40960000u
#define WS_OBSFWD 49152000u
#define WS_ABSBWD 57344000u
#define WS_RBUF   65536000u
#define WS_CBUF   65600000u
#define WS_W16F   65664000u   // fp16 weight arena starts here

// ---- fp16 swizzled weight arena offsets (in halves) ----
// layout per weight: [N/16 tiles][K/32 blocks][32 lanes][16 halves]
#define OFF_COMB16   0
#define OFF_CONV0    32768
#define OFF_CONV1    81920
#define OFF_CONV2    131072
#define OFF_AFW_IH   180224
#define OFF_AFW_HH   229376
#define OFF_ABW_IH   278528
#define OFF_ABW_HH   327680
#define OFF_OFW_IH   376832
#define OFF_OFW_HH   425984
#define OFF_UPD_IH   475136
#define OFF_UPD_HH   536576
#define OFF_POSTVQ   585728
#define OFF_PRIORVQ  618496
#define OFF_ABSFEAT  634880
#define OFF_OBSFEAT  667648
#define OFF_POSTOBS  688128
#define OFF_PRIOROBS 696320

// ---- output float offsets (tuple order, flattened) ----
#define O_REC 0u
#define O_PB  1085824u
#define O_LS  1213568u
#define O_BND 1341568u
#define O_VQ  1469568u
#define O_OH  1469569u
#define O_PAL 2108289u
#define O_POM 2747009u

DEVI float sigmf(float x) { return 1.0f / (1.0f + expf(-x)); }

// convert 8 contiguous f32 -> v8h (two b128 loads + packed cvts)
DEVI v8h f32to8h(const float* p) {
  v4f x = *(const v4f*)p;
  v4f y = *(const v4f*)(p + 4);
  v8h r;
  r[0] = (h16)x[0]; r[1] = (h16)x[1]; r[2] = (h16)x[2]; r[3] = (h16)x[3];
  r[4] = (h16)y[0]; r[5] = (h16)y[1]; r[6] = (h16)y[2]; r[7] = (h16)y[3];
  return r;
}

// A-fragment loaders: base already includes the M-tile row offset
struct AF32 {
  const float* base; int ld;
  DEVI v8h operator()(int mr, int k) const { return f32to8h(base + (size_t)mr * ld + k); }
};
struct AH16 {
  const h16* base; int ld;
  DEVI v8h operator()(int mr, int k) const { return *(const v8h*)(base + (size_t)mr * ld + k); }
};

// ---- wave-level WMMA over one K-segment (nblk 32-wide K blocks).
// A layout per ISA: lane m=lane&15; lane-group g needs K runs [k0+8g, +8) and [k0+16+8g, +8).
// B comes pre-swizzled: lane's 16 halves contiguous.
template <typename AF>
DEVI void wmma_sw(v8f& c, int nblk, AF a, const h16* bseg) {
  const int lane = threadIdx.x & 31;
  const int g8 = (lane >> 4) << 3;
  const int mn = lane & 15;
  for (int kb = 0; kb < nblk; ++kb) {
    const int k0 = kb * 32;
    v8h alo = a(mn, k0 + g8);
    v8h ahi = a(mn, k0 + 16 + g8);
    const h16* bp = bseg + ((size_t)kb * 32 + lane) * 16;
    v8h blo = *(const v8h*)bp;
    v8h bhi = *(const v8h*)(bp + 8);
    v16h av = __builtin_shufflevector(alo, ahi, 0,1,2,3,4,5,6,7,8,9,10,11,12,13,14,15);
    v16h bv = __builtin_shufflevector(blo, bhi, 0,1,2,3,4,5,6,7,8,9,10,11,12,13,14,15);
    c = __builtin_amdgcn_wmma_f32_16x16x32_f16(false, av, false, bv, (short)0, c,
                                               false, false);
  }
}

template <typename SF>
DEVI void wmma_store(const v8f& c, SF st) {
  const int lane = threadIdx.x & 31;
  const int g = (lane >> 4) << 3;
  const int n = lane & 15;
#pragma unroll
  for (int r = 0; r < 8; ++r) st(r + g, n, c[r]);
}

DEVI v8f wmma_bias(const float* bias, int ncolbase) {
  const int n = threadIdx.x & 15;
  float bv = bias ? bias[ncolbase + n] : 0.0f;
  v8f c;
#pragma unroll
  for (int r = 0; r < 8; ++r) c[r] = bv;
  return c;
}

// ---- TDM: async-load one [64][128] f32 step block (8192 elems) into LDS ----
// 6-arg builtin form: (g0, g1, g2, g3, g4, cpol)
DEVI void tdm_load_step(const float* gsrc, unsigned ldsByteAddr) {
  unsigned long long ga = (unsigned long long)(const void*)gsrc;
  u32x4 g0;
  g0[0] = 1u;                                   // count=1, user descriptor
  g0[1] = ldsByteAddr;                          // lds_addr
  g0[2] = (unsigned)ga;                         // global_addr lo
  g0[3] = (unsigned)((ga >> 32) & 0x1FFFFFFu) | (2u << 30);  // addr hi | type=2
  i32x8 g1;
  g1[0] = (int)(2u << 16);                      // data_size = 4 bytes
  g1[1] = (int)(8192u << 16);                   // tensor_dim0 lo16
  g1[2] = (int)((8192u >> 16) | (1u << 16));    // tensor_dim0 hi | tensor_dim1 lo
  g1[3] = (int)(8192u << 16);                   // tile_dim0 = 8192
  g1[4] = 1;                                    // tile_dim1 = 1
  g1[5] = 8192;                                 // tensor_dim0_stride
  g1[6] = 0; g1[7] = 0;
  i32x4 z4; z4[0] = 0; z4[1] = 0; z4[2] = 0; z4[3] = 0;
  i32x8 z8; z8[0] = 0; z8[1] = 0; z8[2] = 0; z8[3] = 0;
  z8[4] = 0; z8[5] = 0; z8[6] = 0; z8[7] = 0;
  __builtin_amdgcn_tensor_load_to_lds(g0, g1, z4, z4, z8, 0);
}

// ======================= weight swizzle kernel =======================
// dst[((nt*KS + ks)*32 + lane)*16 + i] = W[k][nt*16 + (lane&15)]
//   k = ks*32 + 8*(lane>>4) + (i<8 ? i : 8 + i)   (second run offset +16)
__global__ void swz_kernel(const float* __restrict__ w, h16* __restrict__ dst,
                           int K, int N) {
  int i = blockIdx.x * blockDim.x + threadIdx.x;
  if (i >= K * N) return;
  int idx = i & 15;
  int lane = (i >> 4) & 31;
  int rem = i >> 9;
  int KS = K >> 5;
  int ks = rem % KS;
  int nt = rem / KS;
  int g8 = (lane >> 4) << 3;
  int k = ks * 32 + g8 + ((idx < 8) ? idx : (idx + 8));
  int n = nt * 16 + (lane & 15);
  dst[i] = (h16)w[(size_t)k * N + n];
}

// ======================= encoders (tiny K, FMA) =======================
__global__ void encode_kernel(const float* __restrict__ obs, const float* __restrict__ act,
                              const float* __restrict__ encw, const float* __restrict__ encb,
                              const float* __restrict__ actw, const float* __restrict__ actb,
                              float* __restrict__ encobs, float* __restrict__ encact) {
  int i = blockIdx.x * blockDim.x + threadIdx.x;
  if (i >= BT * HD) return;
  int r = i >> 7, hh = i & 127;
  int t = r >> 6, b = r & 63;
  const float* orow = obs + ((size_t)b * TT + t) * 17;
  float s = encb[hh];
#pragma unroll
  for (int k = 0; k < 17; ++k) s += orow[k] * encw[k * HD + hh];
  encobs[i] = s;
  const float* arow = act + ((size_t)b * TT + t) * 6;
  float s2 = actb[hh];
#pragma unroll
  for (int k = 0; k < 6; ++k) s2 += arow[k] * actw[k * HD + hh];
  encact[i] = s2;
}

// ======================= comb / comb_shift (K=256 WMMA) =======================
__global__ __launch_bounds__(512) void comb_kernel(
    const float* __restrict__ encact, const float* __restrict__ encobs,
    const h16* __restrict__ w16, const float* __restrict__ bias,
    float* __restrict__ comb, float* __restrict__ combshift) {
  int wave = threadIdx.x >> 5;
  int gw = blockIdx.x * 16 + wave;
  int mt = gw >> 3, nt = gw & 7;
  int rb = mt * 16, cb = nt * 16;
  const h16* bt = w16 + (size_t)nt * (256 * 16);
  AF32 aAct{encact + (size_t)rb * HD, HD};
  AF32 aObs{encobs + (size_t)rb * HD, HD};
  v8f c = wmma_bias(bias, cb);
  wmma_sw(c, 4, aAct, bt);
  wmma_sw(c, 4, aObs, bt + 4 * 512);
  wmma_store(c, [&](int mr, int nr, float v) { comb[(size_t)(rb + mr) * HD + cb + nr] = v; });
  v8f c2 = wmma_bias(bias, cb);
  if (rb >= 64) {  // t>0: shifted action; t==0 contributes zeros (skip)
    AF32 aShift{encact + (size_t)(rb - 64) * HD, HD};
    wmma_sw(c2, 4, aShift, bt);
  }
  wmma_sw(c2, 4, aObs, bt + 4 * 512);
  wmma_store(c2, [&](int mr, int nr, float v) { combshift[(size_t)(rb + mr) * HD + cb + nr] = v; });
}

// ======================= causal conv (3 taps, segmented K) =======================
__global__ __launch_bounds__(512) void conv_kernel(const float* __restrict__ src,
                                                   float* __restrict__ dst,
                                                   const h16* __restrict__ w16) {
  int wave = threadIdx.x >> 5;
  int gw = blockIdx.x * 16 + wave;
  int mt = gw >> 3, nt = gw & 7;
  int rb = mt * 16, cb = nt * 16;
  const h16* bt = w16 + (size_t)nt * (384 * 16);
  v8f c = wmma_bias(nullptr, cb);
  int tblk = rb >> 6;  // tile-uniform time index
#pragma unroll
  for (int d = 0; d < 3; ++d) {
    if (tblk + d - 2 >= 0) {
      AF32 a{src + (size_t)(rb + 64 * (d - 2)) * HD, HD};
      wmma_sw(c, 4, a, bt + d * 4 * 512);
    }
  }
  wmma_store(c, [&](int mr, int nr, float v) {
    dst[(size_t)(rb + mr) * HD + cb + nr] = fmaxf(v, 0.0f);
  });
}

// ======================= pb_out + log_softmax + hard bnd =======================
__global__ void pbout_kernel(const float* __restrict__ x, const float* __restrict__ w,
                             float* __restrict__ out, float* __restrict__ rbuf,
                             float* __restrict__ cbuf) {
  int r = blockIdx.x * blockDim.x + threadIdx.x;
  if (r >= BT) return;
  const float* xr = x + (size_t)r * HD;
  float l0 = 0.f, l1 = 0.f;
  for (int j = 0; j < HD; ++j) { float v = xr[j]; l0 += v * w[j * 2]; l1 += v * w[j * 2 + 1]; }
  int t = r >> 6, b = r & 63;
  float m = fmaxf(l0, l1);
  float lse = m + logf(expf(l0 - m) + expf(l1 - m));
  size_t ob = ((size_t)b * TT + t) * 2;
  out[O_LS + ob] = l0 - lse;
  out[O_LS + ob + 1] = l1 - lse;
  float h0 = (l0 >= l1) ? 1.0f : 0.0f;
  float h1 = 1.0f - h0;
  if (t <= 1 || t >= TT - 1) { h0 = 1.0f; h1 = 0.0f; }
  out[O_BND + ob] = h0;
  out[O_BND + ob + 1] = h1;
  rbuf[r] = h0;
  cbuf[r] = h1;
}

// ======================= three GRU scans, TDM double-buffered x =======================
__global__ __launch_bounds__(1024) void gru3_kernel(
    const float* __restrict__ comb, const float* __restrict__ encobs,
    const float* __restrict__ cbuf,
    const h16* afw_ih, const h16* afw_hh, const float* afw_bih, const float* afw_bhh,
    const h16* abw_ih, const h16* abw_hh, const float* abw_bih, const float* abw_bhh,
    const h16* ofw_ih, const h16* ofw_hh, const float* ofw_bih, const float* ofw_bhh,
    float* __restrict__ absfwd, float* __restrict__ obsfwd, float* __restrict__ absbwd) {
  extern __shared__ char smemraw[];
  float* sm   = (float*)smemraw;
  float* h32  = sm;                  // 8192 f
  float* gsum = sm + 8192;           // 16384 f
  float* gxn  = sm + 24576;          // 8192 f
  float* ghn  = sm + 32768;          // 8192 f
  float* xst  = sm + 40960;          // 2 x 8192 f TDM staging
  h16*  x16   = (h16*)(sm + 57344);  // 8192 h
  h16*  hs16  = x16 + 8192;          // 8192 h

  const int mode = blockIdx.x;  // 0: abs fwd, 1: obs fwd (cp-gated), 2: abs bwd (reverse)
  const float* xin = (mode == 1) ? encobs : comb;
  const h16* wih = (mode == 0) ? afw_ih : (mode == 1) ? ofw_ih : abw_ih;
  const h16* whh = (mode == 0) ? afw_hh : (mode == 1) ? ofw_hh : abw_hh;
  const float* bih = (mode == 0) ? afw_bih : (mode == 1) ? ofw_bih : abw_bih;
  const float* bhh = (mode == 0) ? afw_bhh : (mode == 1) ? ofw_bhh : abw_bhh;
  float* hist = (mode == 0) ? absfwd : (mode == 1) ? obsfwd : absbwd;
  const int rev = (mode == 2);

  const int tid = threadIdx.x, wave = tid >> 5;
  for (int e = tid; e < 8192; e += 1024) h32[e] = 0.0f;
  int cur = 0;
  if (wave == 0)
    tdm_load_step(xin + (size_t)(rev ? TT - 1 : 0) * 8192, 163840u /*xst byte off*/);
  __syncthreads();

  for (int s = 0; s < TT; ++s) {
    const int t = rev ? (TT - 1 - s) : s;
    const size_t base = (size_t)t * 8192;
    if (wave == 0) __builtin_amdgcn_s_wait_tensorcnt(0);  // stage[cur] ready
    __syncthreads();
    const int tn = rev ? (t - 1) : (t + 1);
    if (wave == 0 && tn >= 0 && tn < TT)
      tdm_load_step(xin + (size_t)tn * 8192, 163840u + (unsigned)((1 ^ cur) * 32768));
    const float* xcur = xst + cur * 8192;
    for (int e = tid; e < 8192; e += 1024) {
      x16[e] = (h16)xcur[e];
      float hv = h32[e];
      if (mode == 1) hv *= cbuf[t * 64 + (e >> 7)];
      h32[e] = hv;
      hs16[e] = (h16)hv;
    }
    __syncthreads();
    for (int j = wave; j < 128; j += 32) {
      if (j < 64) {  // r,z gates: x@Wih + h@Whh summed
        int mt = j & 3, nt = j >> 2, rb = mt * 16, cb = nt * 16;
        v8f c;
        { int n = tid & 15; float bv = bih[cb + n] + bhh[cb + n];
#pragma unroll
          for (int r2 = 0; r2 < 8; ++r2) c[r2] = bv; }
        wmma_sw(c, 4, AH16{x16 + (size_t)rb * HD, HD}, wih + (size_t)nt * 2048);
        wmma_sw(c, 4, AH16{hs16 + (size_t)rb * HD, HD}, whh + (size_t)nt * 2048);
        wmma_store(c, [&](int mr, int nr, float v) { gsum[(rb + mr) * 256 + cb + nr] = v; });
      } else {       // n gate halves kept separate (xn, hn)
        int j2 = j - 64, sel = j2 >> 5, mt = j2 & 3, nt8 = (j2 >> 2) & 7;
        int rb = mt * 16, cb = 256 + nt8 * 16, cl = nt8 * 16;
        const float* bb = sel ? bhh : bih;
        v8f c;
        { int n = tid & 15; float bv = bb[cb + n];
#pragma unroll
          for (int r2 = 0; r2 < 8; ++r2) c[r2] = bv; }
        if (!sel) {
          wmma_sw(c, 4, AH16{x16 + (size_t)rb * HD, HD}, wih + (size_t)(cb >> 4) * 2048);
          wmma_store(c, [&](int mr, int nr, float v) { gxn[(rb + mr) * HD + cl + nr] = v; });
        } else {
          wmma_sw(c, 4, AH16{hs16 + (size_t)rb * HD, HD}, whh + (size_t)(cb >> 4) * 2048);
          wmma_store(c, [&](int mr, int nr, float v) { ghn[(rb + mr) * HD + cl + nr] = v; });
        }
      }
    }
    __syncthreads();
    for (int e = tid; e < 8192; e += 1024) {
      int b2 = e >> 7, jj = e & 127;
      float r2 = sigmf(gsum[b2 * 256 + jj]);
      float z  = sigmf(gsum[b2 * 256 + 128 + jj]);
      float nn = tanhf(gxn[e] + r2 * ghn[e]);
      float hn = (1.0f - z) * nn + z * h32[e];
      h32[e] = hn;
      hist[base + e] = hn;
    }
    cur ^= 1;
    __syncthreads();
  }
}

// ======================= main 998-step stateful scan (single WGP) =======================
__global__ __launch_bounds__(1024) void main_scan_kernel(
    const float* __restrict__ absfwd, const float* __restrict__ absbwd,
    const float* __restrict__ obsfwd,
    const float* __restrict__ rbuf, const float* __restrict__ cbuf,
    const h16* w_postvq, const float* b_postvq,
    const float* __restrict__ codebook, const float* __restrict__ prior_codebook,
    const h16* w_priorvq, const float* b_priorvq,
    const h16* w_absfeat, const float* b_absfeat,
    const h16* w_obsfeat, const float* b_obsfeat,
    const h16* w_postobs, const float* b_postobs,
    const h16* w_priorobs, const float* b_priorobs,
    const h16* upd_ih, const h16* upd_hh, const float* upd_bih, const float* upd_bhh,
    const float* __restrict__ pbw, const float* __restrict__ pbb,
    float* __restrict__ out, float* __restrict__ ofeat_ws) {
  extern __shared__ char smemraw[];
  float* sm      = (float*)smemraw;
  float* absbel  = sm;              // 8192
  float* absst   = sm + 8192;       // 8192
  float* obsbel  = sm + 16384;      // 8192
  float* obsst   = sm + 24576;      // 2048
  float* absfeat = sm + 26624;      // 8192
  float* scratch = sm + 34816;      // 32768 (stage-overlaid)
  float* dists   = sm + 67584;      // 640
  float* rdv     = sm + 68224;      // 64
  float* cpv     = sm + 68288;      // 64
  float* s_vq    = sm + 68352;
  float* s_acc   = sm + 68353;
  int*   idxs    = (int*)(sm + 68354);  // 64 ints
  float* feat  = scratch;           // stages B..C
  float* gsum  = scratch;           // stage E
  float* gxn   = scratch + 16384;
  float* ghn   = scratch + 24576;
  float* pf    = scratch;           // stage F
  float* ofeat = scratch + 8192;    // stage I..J

  const int tid = threadIdx.x, wave = tid >> 5;
  for (int e = tid; e < 8192; e += 1024) { absbel[e] = 0; absst[e] = 0; obsbel[e] = 0; }
  for (int e = tid; e < 2048; e += 1024) obsst[e] = 0;
  if (tid == 0) *s_acc = 0.0f;
  __syncthreads();

  for (int tp = 0; tp < TPN; ++tp) {
    const int t = tp + 1;
    const size_t ga = (size_t)tp * 8192;  // abs_fwd[t-1]
    const size_t gb = (size_t)t * 8192;   // abs_bwd[t]
    const size_t go = (size_t)t * 8192;   // obs_fwd[t]
    // ---- stage A: abs_belief update + next-step prefetch ----
    if (tid < 64) { rdv[tid] = rbuf[t * 64 + tid]; cpv[tid] = cbuf[t * 64 + tid]; }
    if (tid == 0) *s_vq = 0.0f;
    if (tid < 256 && tp + 1 < TPN) {  // prefetch next step's history rows
      __builtin_prefetch(absfwd + ga + 8192 + tid * 32, 0, 0);
      __builtin_prefetch(absbwd + gb + 8192 + tid * 32, 0, 0);
      __builtin_prefetch(obsfwd + go + 8192 + tid * 32, 0, 0);
    }
    __syncthreads();
    for (int e = tid; e < 8192; e += 1024) {
      int b2 = e >> 7;
      absbel[e] = rdv[b2] * absfwd[ga + e] + cpv[b2] * absbel[e];
    }
    __syncthreads();
    // ---- stage B: feat = post_vq([afp|abt]) ----
    {
      int mt = wave & 3, nt = wave >> 2, rb = mt * 16, cb = nt * 16;
      const h16* bt = w_postvq + (size_t)nt * 4096;
      v8f c = wmma_bias(b_postvq, cb);
      wmma_sw(c, 4, AF32{absfwd + ga + (size_t)rb * HD, HD}, bt);
      wmma_sw(c, 4, AF32{absbwd + gb + (size_t)rb * HD, HD}, bt + 2048);
      wmma_store(c, [&](int mr, int nr, float v) { feat[(rb + mr) * HD + cb + nr] = v; });
    }
    __syncthreads();
    // ---- stage C: VQ argmin, one-hot, vq loss, abs_state update ----
    if (tid < 640) {
      int b2 = tid / 10, k = tid % 10;
      float s = 0.f;
      for (int j = 0; j < HD; ++j) { float d = feat[b2 * HD + j] - codebook[k * HD + j]; s += d * d; }
      dists[tid] = s;
    }
    __syncthreads();
    if (tid < 64) {
      float best = dists[tid * 10]; int bi = 0;
      for (int k = 1; k < KCB; ++k) { float d = dists[tid * 10 + k]; if (d < best) { best = d; bi = k; } }
      idxs[tid] = bi;
    }
    __syncthreads();
    if (tid < 640) {
      int b2 = tid / 10, k = tid % 10;
      out[O_OH + ((size_t)b2 * TPN + tp) * KCB + k] = (k == idxs[b2]) ? 1.0f : 0.0f;
    }
    {
      float loc = 0.f;
      for (int e = tid; e < 8192; e += 1024) {
        int b2 = e >> 7, jj = e & 127;
        float cbv = codebook[idxs[b2] * HD + jj];
        float d = feat[e] - cbv;
        loc += d * d;
        absst[e] = rdv[b2] * cbv + cpv[b2] * absst[e];
      }
      atomicAdd(s_vq, loc);
    }
    __syncthreads();
    if (tid == 0) *s_acc += 1.25f * (*s_vq) / 8192.0f;
    // ---- stage D: abs_feat = lin([abs_belief|abs_state]) ----
    {
      int mt = wave & 3, nt = wave >> 2, rb = mt * 16, cb = nt * 16;
      const h16* bt = w_absfeat + (size_t)nt * 4096;
      v8f c = wmma_bias(b_absfeat, cb);
      wmma_sw(c, 4, AF32{absbel + (size_t)rb * HD, HD}, bt);
      wmma_sw(c, 4, AF32{absst + (size_t)rb * HD, HD}, bt + 2048);
      wmma_store(c, [&](int mr, int nr, float v) { absfeat[(rb + mr) * HD + cb + nr] = v; });
    }
    __syncthreads();
    // ---- stage E: update_obs_belief GRU (x=[obs_state|abs_feat] K=160) ----
    for (int j = wave; j < 128; j += 32) {
      if (j < 64) {
        int mt = j & 3, nt = j >> 2, rb = mt * 16, cb = nt * 16;
        const h16* btx = upd_ih + (size_t)nt * 2560;
        const h16* bth = upd_hh + (size_t)nt * 2048;
        v8f c;
        { int n = tid & 15; float bv = upd_bih[cb + n] + upd_bhh[cb + n];
#pragma unroll
          for (int r2 = 0; r2 < 8; ++r2) c[r2] = bv; }
        wmma_sw(c, 1, AF32{obsst + (size_t)rb * SD, SD}, btx);
        wmma_sw(c, 4, AF32{absfeat + (size_t)rb * HD, HD}, btx + 512);
        wmma_sw(c, 4, AF32{obsbel + (size_t)rb * HD, HD}, bth);
        wmma_store(c, [&](int mr, int nr, float v) { gsum[(rb + mr) * 256 + cb + nr] = v; });
      } else {
        int j2 = j - 64, sel = j2 >> 5, mt = j2 & 3, nt8 = (j2 >> 2) & 7;
        int rb = mt * 16, cb = 256 + nt8 * 16, cl = nt8 * 16;
        const float* bb = sel ? upd_bhh : upd_bih;
        v8f c;
        { int n = tid & 15; float bv = bb[cb + n];
#pragma unroll
          for (int r2 = 0; r2 < 8; ++r2) c[r2] = bv; }
        if (!sel) {
          const h16* btx = upd_ih + (size_t)(cb >> 4) * 2560;
          wmma_sw(c, 1, AF32{obsst + (size_t)rb * SD, SD}, btx);
          wmma_sw(c, 4, AF32{absfeat + (size_t)rb * HD, HD}, btx + 512);
          wmma_store(c, [&](int mr, int nr, float v) { gxn[(rb + mr) * HD + cl + nr] = v; });
        } else {
          wmma_sw(c, 4, AF32{obsbel + (size_t)rb * HD, HD}, upd_hh + (size_t)(cb >> 4) * 2048);
          wmma_store(c, [&](int mr, int nr, float v) { ghn[(rb + mr) * HD + cl + nr] = v; });
        }
      }
    }
    __syncthreads();
    for (int e = tid; e < 8192; e += 1024) {
      int b2 = e >> 7, jj = e & 127;
      float r2 = sigmf(gsum[b2 * 256 + jj]);
      float z  = sigmf(gsum[b2 * 256 + 128 + jj]);
      float nn = tanhf(gxn[e] + r2 * ghn[e]);
      float upd = (1.0f - z) * nn + z * obsbel[e];
      obsbel[e] = rdv[b2] * absfeat[e] + cpv[b2] * upd;
    }
    __syncthreads();
    // ---- stage F: prior_vq feat, prior_obs_mean (out), new obs_state ----
    for (int j = wave; j < 48; j += 32) {
      if (j < 32) {
        int mt = j & 3, nt = j >> 2, rb = mt * 16, cb = nt * 16;
        v8f c = wmma_bias(b_priorvq, cb);
        wmma_sw(c, 4, AF32{absbel + (size_t)rb * HD, HD}, w_priorvq + (size_t)nt * 2048);
        wmma_store(c, [&](int mr, int nr, float v) { pf[(rb + mr) * HD + cb + nr] = v; });
      } else if (j < 40) {
        int j2 = j - 32, mt = j2 & 3, nt = j2 >> 2, rb = mt * 16, cb = nt * 16;
        v8f c = wmma_bias(b_priorobs, cb);
        wmma_sw(c, 4, AF32{obsbel + (size_t)rb * HD, HD}, w_priorobs + (size_t)nt * 2048);
        wmma_store(c, [&](int mr, int nr, float v) {
          out[O_POM + ((size_t)(rb + mr) * TPN + tp) * SD + cb + nr] = v;
        });
      } else {
        int j2 = j - 40, mt = j2 & 3, nt = j2 >> 2, rb = mt * 16, cb = nt * 16;
        const h16* bt = w_postobs + (size_t)nt * 4096;
        v8f c = wmma_bias(b_postobs, cb);
        wmma_sw(c, 4, AF32{obsfwd + go + (size_t)rb * HD, HD}, bt);
        wmma_sw(c, 4, AF32{absfeat + (size_t)rb * HD, HD}, bt + 2048);
        wmma_store(c, [&](int mr, int nr, float v) { obsst[(rb + mr) * SD + cb + nr] = v; });
      }
    }
    __syncthreads();
    // ---- stage F2 + I: prior_abs_logits (from pf), obs_feat ----
    if (tid < 640) {
      int b2 = tid / 10, k = tid % 10;
      float s = 0.f;
      for (int j = 0; j < HD; ++j) {
        float d = pf[b2 * HD + j] - prior_codebook[k * HD + j];
        s += d * d;
      }
      out[O_PAL + ((size_t)b2 * TPN + tp) * KCB + k] = -s;
    }
    {
      int mt = wave & 3, nt = wave >> 2, rb = mt * 16, cb = nt * 16;
      const h16* bt = w_obsfeat + (size_t)nt * 2560;
      v8f c = wmma_bias(b_obsfeat, cb);
      wmma_sw(c, 4, AF32{obsbel + (size_t)rb * HD, HD}, bt);
      wmma_sw(c, 1, AF32{obsst + (size_t)rb * SD, SD}, bt + 2048);
      wmma_store(c, [&](int mr, int nr, float v) {
        int r2 = rb + mr;
        ofeat[r2 * HD + cb + nr] = v;
        ofeat_ws[(size_t)tp * 8192 + r2 * HD + cb + nr] = v;
      });
    }
    __syncthreads();
    // ---- stage J: prior_boundary (2 cols) ----
    if (tid < 128) {
      int b2 = tid >> 1, c2 = tid & 1;
      float s = pbb[c2];
      for (int j = 0; j < HD; ++j) s += ofeat[b2 * HD + j] * pbw[j * 2 + c2];
      out[O_PB + ((size_t)b2 * TPN + tp) * 2 + c2] = s;
    }
    __syncthreads();
  }
  if (tid == 0) out[O_VQ] = *s_acc / (float)TPN;
}

// ======================= decoder =======================
__global__ void dec_kernel(const float* __restrict__ ofeat_ws, const float* __restrict__ decw,
                           const float* __restrict__ decb, float* __restrict__ out) {
  int i = blockIdx.x * blockDim.x + threadIdx.x;
  const int n = BATCH * TPN * 17;
  if (i >= n) return;
  int o = i % 17, bt = i / 17, b2 = bt / TPN, tp = bt % TPN;
  const float* xr = ofeat_ws + ((size_t)tp * 64 + b2) * HD;
  float s = decb[o];
  for (int j = 0; j < HD; ++j) s += xr[j] * decw[j * 17 + o];
  out[O_REC + i] = s;
}

// ======================= launch =======================
extern "C" void kernel_launch(void* const* d_in, const int* in_sizes, int n_in,
                              void* d_out, int out_size, void* d_ws, size_t ws_size,
                              hipStream_t stream) {
  (void)in_sizes; (void)n_in; (void)out_size; (void)ws_size;
  float* ws = (float*)d_ws;
  h16* w16 = (h16*)(ws + WS_W16F);
  float* out = (float*)d_out;
  const float* obs = (const float*)d_in[0];
  const float* act = (const float*)d_in[1];
  #define FIN(i) ((const float*)d_in[i])

  auto swz = [&](int idx, int off, int K, int N) {
    swz_kernel<<<(K * N + 255) / 256, 256, 0, stream>>>(FIN(idx), w16 + off, K, N);
  };
  swz(6, OFF_COMB16, 256, 128);
  swz(8, OFF_CONV0, 384, 128);  swz(9, OFF_CONV1, 384, 128);  swz(10, OFF_CONV2, 384, 128);
  swz(12, OFF_AFW_IH, 128, 384); swz(13, OFF_AFW_HH, 128, 384);
  swz(16, OFF_ABW_IH, 128, 384); swz(17, OFF_ABW_HH, 128, 384);
  swz(20, OFF_OFW_IH, 128, 384); swz(21, OFF_OFW_HH, 128, 384);
  swz(24, OFF_UPD_IH, 160, 384); swz(25, OFF_UPD_HH, 128, 384);
  swz(28, OFF_POSTVQ, 256, 128);
  swz(31, OFF_PRIORVQ, 128, 128);
  swz(34, OFF_ABSFEAT, 256, 128);
  swz(36, OFF_OBSFEAT, 160, 128);
  swz(38, OFF_POSTOBS, 256, 32);
  swz(40, OFF_PRIOROBS, 128, 32);

  encode_kernel<<<(BT * HD + 255) / 256, 256, 0, stream>>>(
      obs, act, FIN(2), FIN(3), FIN(4), FIN(5), ws + WS_ENCOBS, ws + WS_ENCACT);
  comb_kernel<<<2000, 512, 0, stream>>>(ws + WS_ENCACT, ws + WS_ENCOBS,
                                        w16 + OFF_COMB16, FIN(7),
                                        ws + WS_COMB, ws + WS_CBUFA);
  conv_kernel<<<2000, 512, 0, stream>>>(ws + WS_CBUFA, ws + WS_CBUFB, w16 + OFF_CONV0);
  conv_kernel<<<2000, 512, 0, stream>>>(ws + WS_CBUFB, ws + WS_CBUFA, w16 + OFF_CONV1);
  conv_kernel<<<2000, 512, 0, stream>>>(ws + WS_CBUFA, ws + WS_CBUFB, w16 + OFF_CONV2);
  pbout_kernel<<<(BT + 255) / 256, 256, 0, stream>>>(ws + WS_CBUFB, FIN(11), out,
                                                     ws + WS_RBUF, ws + WS_CBUF);
  gru3_kernel<<<3, 1024, 262144, stream>>>(
      ws + WS_COMB, ws + WS_ENCOBS, ws + WS_CBUF,
      w16 + OFF_AFW_IH, w16 + OFF_AFW_HH, FIN(14), FIN(15),
      w16 + OFF_ABW_IH, w16 + OFF_ABW_HH, FIN(18), FIN(19),
      w16 + OFF_OFW_IH, w16 + OFF_OFW_HH, FIN(22), FIN(23),
      ws + WS_ABSFWD, ws + WS_OBSFWD, ws + WS_ABSBWD);
  main_scan_kernel<<<1, 1024, 273920, stream>>>(
      ws + WS_ABSFWD, ws + WS_ABSBWD, ws + WS_OBSFWD,
      ws + WS_RBUF, ws + WS_CBUF,
      w16 + OFF_POSTVQ, FIN(29), FIN(30), FIN(33),
      w16 + OFF_PRIORVQ, FIN(32),
      w16 + OFF_ABSFEAT, FIN(35),
      w16 + OFF_OBSFEAT, FIN(37),
      w16 + OFF_POSTOBS, FIN(39),
      w16 + OFF_PRIOROBS, FIN(41),
      w16 + OFF_UPD_IH, w16 + OFF_UPD_HH, FIN(26), FIN(27),
      FIN(42), FIN(43), out, ws + WS_ENCACT /* reused as obs_feat_seq */);
  dec_kernel<<<(BATCH * TPN * 17 + 255) / 256, 256, 0, stream>>>(
      ws + WS_ENCACT, FIN(44), FIN(45), out);
  #undef FIN
}
